// SQLDecoder_77266461655077
// MI455X (gfx1250) — matrix operand order repaired and checked by reading, not compiled, not run on hardware
//
#include <hip/hip_runtime.h>
#include <math.h>

typedef __bf16 bf16;
typedef __attribute__((ext_vector_type(16))) __bf16 bf16x16;
typedef __attribute__((ext_vector_type(8)))  __bf16 bf16x8;
typedef __attribute__((ext_vector_type(8)))  float  f32x8;

#define B_   256
#define S_   512
#define E_   1024
#define H_   1024
#define AV_  1024
#define A_   128
#define TY_  128
#define T_   48
#define NP   97
#define IN_  1280
#define KCAT 2304   // IN_ + H_

// ---------------------------------------------------------------- converts
__global__ void k_cvt(const float* __restrict__ in, bf16* __restrict__ out, long n) {
  long i = (long)blockIdx.x * blockDim.x + threadIdx.x;
  long st = (long)gridDim.x * blockDim.x;
  for (; i < n; i += st) out[i] = (bf16)in[i];
}

__global__ void k_fill_f32(float* p, float v, long n) {
  long i = (long)blockIdx.x * blockDim.x + threadIdx.x;
  long st = (long)gridDim.x * blockDim.x;
  for (; i < n; i += st) p[i] = v;
}

__global__ void k_fill_bf16(bf16* p, float v, long n) {
  long i = (long)blockIdx.x * blockDim.x + threadIdx.x;
  long st = (long)gridDim.x * blockDim.x;
  for (; i < n; i += st) p[i] = (bf16)v;
}

__global__ void k_build_wcat(const float* __restrict__ wih, const float* __restrict__ whh,
                             bf16* __restrict__ wcat) {
  long n = (long)4 * H_ * KCAT;
  long i = (long)blockIdx.x * blockDim.x + threadIdx.x;
  long st = (long)gridDim.x * blockDim.x;
  for (; i < n; i += st) {
    int row = (int)(i / KCAT), col = (int)(i % KCAT);
    float v = (col < IN_) ? wih[(size_t)row * IN_ + col]
                          : whh[(size_t)row * H_ + (col - IN_)];
    wcat[i] = (bf16)v;
  }
}

// ---------------------------------------------------------------- WMMA GEMM
// C[M,N] = A[M,K] * Bw[N,K]^T ; A,Bw bf16 row-major; C fp32 or bf16.
// M,N multiples of 128; K multiple of 32 (K >= 64 assumed for pipeline; all call sites comply).
// Double-buffered LDS pipeline fed by GLOBAL_LOAD_ASYNC_TO_LDS_B128 (ASYNCcnt),
// last K-slice peeled so the steady-state loop is branchless.
#define LDSS 40  // padded LDS row stride (elements), 80B rows keep 16B alignment
__global__ __launch_bounds__(256) void k_gemm_bf16(
    const bf16* __restrict__ A, const bf16* __restrict__ Bw,
    void* __restrict__ Cout, int M, int N, int K, int out_bf16) {
  __shared__ __align__(16) bf16 Asm[2][128 * LDSS];
  __shared__ __align__(16) bf16 Bsm[2][128 * LDSS];

  const int tid  = threadIdx.x;
  const int lane = tid & 31;
  const int w    = tid >> 5;          // 8 waves
  const int wr   = w & 3;             // 4 row-blocks of 32
  const int wc   = w >> 2;            // 2 col-blocks of 64
  const int m0   = blockIdx.y * 128;
  const int n0   = blockIdx.x * 128;

  f32x8 acc[2][4] = {};

  const int lrow = tid >> 1;          // 0..127
  const int lcol = (tid & 1) * 16;    // 0 or 16
  const int lm   = lane & 15;
  const int khalf = lane >> 4;

  // Per-thread source rows (advance by 32 elements per K-step; +16B handled via offset:16)
  const bf16* gA = A  + (size_t)(m0 + lrow) * K + lcol;
  const bf16* gB = Bw + (size_t)(n0 + lrow) * K + lcol;
  const unsigned ldsA0 = (unsigned)(uintptr_t)&Asm[0][lrow * LDSS + lcol];
  const unsigned ldsA1 = (unsigned)(uintptr_t)&Asm[1][lrow * LDSS + lcol];
  const unsigned ldsB0 = (unsigned)(uintptr_t)&Bsm[0][lrow * LDSS + lcol];
  const unsigned ldsB1 = (unsigned)(uintptr_t)&Bsm[1][lrow * LDSS + lcol];

  // Issue 4 async B128 global->LDS copies (A lo/hi, B lo/hi) for K-slice k0 into buffer `buf`.
  auto issue = [&](int buf, int k0) {
    unsigned la = buf ? ldsA1 : ldsA0;
    unsigned lb = buf ? ldsB1 : ldsB0;
    unsigned long long aa = (unsigned long long)(uintptr_t)(gA + k0);
    unsigned long long bb = (unsigned long long)(uintptr_t)(gB + k0);
    asm volatile(
        "global_load_async_to_lds_b128 %0, %2, off\n\t"
        "global_load_async_to_lds_b128 %0, %2, off offset:16\n\t"
        "global_load_async_to_lds_b128 %1, %3, off\n\t"
        "global_load_async_to_lds_b128 %1, %3, off offset:16"
        :: "v"(la), "v"(lb), "v"(aa), "v"(bb)
        : "memory");
  };

  // 8 WMMAs on LDS buffer `buf`.
  auto compute = [&](int buf) {
    // A fragments: lanes 0-15 row M=lm hold K {0..7,16..23}; lanes 16-31 K {8..15,24..31}
    union { bf16x16 v; bf16x8 h[2]; } fa[2], fb[4];
#pragma unroll
    for (int i = 0; i < 2; i++) {
      int r = wr * 32 + i * 16 + lm;
      fa[i].h[0] = *(const bf16x8*)&Asm[buf][r * LDSS + khalf * 8];
      fa[i].h[1] = *(const bf16x8*)&Asm[buf][r * LDSS + 16 + khalf * 8];
    }
  // B fragments: lane holds column N=lm, same K order as A: {0..7,16..23} / {8..15,24..31}
#pragma unroll
    for (int j = 0; j < 4; j++) {
      int r = wc * 64 + j * 16 + lm;
      fb[j].h[0] = *(const bf16x8*)&Bsm[buf][r * LDSS + khalf * 8];
      fb[j].h[1] = *(const bf16x8*)&Bsm[buf][r * LDSS + 16 + khalf * 8];
    }
#pragma unroll
    for (int i = 0; i < 2; i++)
#pragma unroll
      for (int j = 0; j < 4; j++)
        acc[i][j] = __builtin_amdgcn_wmma_f32_16x16x32_bf16(
            false, fa[i].v, false, fb[j].v, (short)0, acc[i][j], false, false);
  };

  issue(0, 0);
  int cur = 0;
  int k0 = 0;
  // steady state: branchless issue/wait/compute
  for (; k0 + 32 < K; k0 += 32) {
    issue(cur ^ 1, k0 + 32);                           // overlap next tile with compute
    asm volatile("s_wait_asynccnt 0x4" ::: "memory");  // in-order: oldest 4 (cur) landed
    __syncthreads();
    compute(cur);
    __syncthreads();   // all waves done reading buf[cur] before it is refilled
    cur ^= 1;
  }
  // epilogue: last K-slice
  asm volatile("s_wait_asynccnt 0x0" ::: "memory");
  __syncthreads();
  compute(cur);

  // C layout: lane L, VGPR r -> (M = r + 8*(L/16), N = L%16)
#pragma unroll
  for (int i = 0; i < 2; i++)
#pragma unroll
    for (int j = 0; j < 4; j++)
#pragma unroll
      for (int r = 0; r < 8; r++) {
        int mm = m0 + wr * 32 + i * 16 + khalf * 8 + r;
        int nn = n0 + wc * 64 + j * 16 + lm;
        float v = acc[i][j][r];
        if (out_bf16) ((bf16*)Cout)[(size_t)mm * N + nn] = (bf16)v;
        else          ((float*)Cout)[(size_t)mm * N + nn] = v;
      }
}

// ---------------------------------------------------------------- init
__global__ void k_pool_max(const float* __restrict__ enc, bf16* __restrict__ pooled) {
  int idx = blockIdx.x * blockDim.x + threadIdx.x;   // b*E + e
  if (idx >= B_ * E_) return;
  int b = idx >> 10, e = idx & 1023;
  const float* p = enc + (size_t)b * S_ * E_ + e;
  float m = p[0];
  for (int s = 1; s < S_; s++) m = fmaxf(m, p[(size_t)s * E_]);
  pooled[idx] = (bf16)m;
}

__global__ void k_bias_tanh(const float* __restrict__ in, const float* __restrict__ bias,
                            float* __restrict__ outf, bf16* __restrict__ outb,
                            int rows, int cols) {
  int idx = blockIdx.x * blockDim.x + threadIdx.x;
  if (idx >= rows * cols) return;
  int c = idx % cols;
  float v = in[idx] + (bias ? bias[c] : 0.f);
  v = tanhf(v);
  if (outf) outf[idx] = v;
  if (outb) outb[idx] = (bf16)v;
}

// ---------------------------------------------------------------- step kernels
__global__ void k_build_x(const int* __restrict__ act, const int* __restrict__ typ,
                          const float* __restrict__ prod_embed,
                          const float* __restrict__ type_tab,
                          const bf16* __restrict__ att, const bf16* __restrict__ hbf,
                          bf16* __restrict__ x, float mval) {
  int idx = blockIdx.x * blockDim.x + threadIdx.x;   // b*KCAT + col
  if (idx >= B_ * KCAT) return;
  int b = idx / KCAT, col = idx % KCAT;
  if (col >= IN_) { x[idx] = hbf[b * H_ + (col - IN_)]; return; }
  float v;
  if (col < A_)            v = prod_embed[act[b] * A_ + col] * mval;
  else if (col < A_ + AV_) v = (float)att[b * AV_ + (col - A_)] * mval;
  else                     v = type_tab[typ[b] * TY_ + (col - A_ - AV_)] * mval;
  x[idx] = (bf16)v;
}

__global__ void k_lstm(const float* __restrict__ gates, const float* __restrict__ bih,
                       const float* __restrict__ bhh, float* __restrict__ c,
                       bf16* __restrict__ hbf) {
  int idx = blockIdx.x * blockDim.x + threadIdx.x;
  if (idx >= B_ * H_) return;
  int b = idx >> 10, j = idx & 1023;
  const float* g = gates + (size_t)b * 4 * H_;
  float gi = g[j]            + bih[j]            + bhh[j];
  float gf = g[H_ + j]       + bih[H_ + j]       + bhh[H_ + j];
  float gg = g[2 * H_ + j]   + bih[2 * H_ + j]   + bhh[2 * H_ + j];
  float go = g[3 * H_ + j]   + bih[3 * H_ + j]   + bhh[3 * H_ + j];
  float si = 1.f / (1.f + expf(-gi));
  float sf = 1.f / (1.f + expf(-gf));
  float so = 1.f / (1.f + expf(-go));
  float ct = sf * c[idx] + si * tanhf(gg);
  float ht = so * tanhf(ct);
  c[idx] = ct;
  hbf[idx] = (bf16)ht;
}

__global__ __launch_bounds__(256) void k_scores(const bf16* __restrict__ srclin,
                                                const bf16* __restrict__ hbf,
                                                float* __restrict__ scores) {
  __shared__ __align__(16) bf16 hs[H_];
  int b = blockIdx.y;
  int s = blockIdx.x * 256 + threadIdx.x;
  for (int k = threadIdx.x; k < H_; k += 256) hs[k] = hbf[b * H_ + k];
  __syncthreads();
  const bf16* row = srclin + ((size_t)b * S_ + s) * H_;
  float acc = 0.f;
  for (int k = 0; k < H_; k += 8) {
    bf16x8 a = *(const bf16x8*)(row + k);
    bf16x8 h = *(const bf16x8*)&hs[k];
#pragma unroll
    for (int u = 0; u < 8; u++) acc += (float)a[u] * (float)h[u];
  }
  scores[b * S_ + s] = acc;
}

__global__ __launch_bounds__(256) void k_softmax(const float* __restrict__ sc,
                                                 float* __restrict__ w) {
  __shared__ float red[256];
  int b = blockIdx.x, t = threadIdx.x;
  float v0 = sc[b * S_ + t], v1 = sc[b * S_ + t + 256];
  red[t] = fmaxf(v0, v1);
  __syncthreads();
  for (int o = 128; o > 0; o >>= 1) { if (t < o) red[t] = fmaxf(red[t], red[t + o]); __syncthreads(); }
  float mx = red[0];
  __syncthreads();
  float e0 = expf(v0 - mx), e1 = expf(v1 - mx);
  red[t] = e0 + e1;
  __syncthreads();
  for (int o = 128; o > 0; o >>= 1) { if (t < o) red[t] += red[t + o]; __syncthreads(); }
  float inv = 1.f / red[0];
  w[b * S_ + t] = e0 * inv;
  w[b * S_ + t + 256] = e1 * inv;
}

__global__ __launch_bounds__(256) void k_ctx_hc(const float* __restrict__ w,
                                                const bf16* __restrict__ src,
                                                const bf16* __restrict__ hbf,
                                                bf16* __restrict__ hc) {
  __shared__ float ws[S_];
  int b = blockIdx.x, t = threadIdx.x;
  ws[t] = w[b * S_ + t];
  ws[t + 256] = w[b * S_ + t + 256];
  for (int k = t; k < H_; k += 256) hc[(size_t)b * 2 * H_ + k] = hbf[b * H_ + k];
  __syncthreads();
  float acc[4] = {0.f, 0.f, 0.f, 0.f};
  const bf16* base = src + (size_t)b * S_ * E_;
  for (int s = 0; s < S_; s++) {
    float wv = ws[s];
    const bf16* rp = base + (size_t)s * E_;
#pragma unroll
    for (int u = 0; u < 4; u++) acc[u] += wv * (float)rp[t + u * 256];
  }
#pragma unroll
  for (int u = 0; u < 4; u++)
    hc[(size_t)b * 2 * H_ + H_ + t + u * 256] = (bf16)acc[u];
}

__global__ __launch_bounds__(128) void k_readout(const float* __restrict__ q,
                                                 const float* __restrict__ pe,
                                                 const float* __restrict__ pb,
                                                 float* __restrict__ out) {
  __shared__ float qs[A_];
  __shared__ float lg[NP];
  __shared__ float mx, sm;
  int b = blockIdx.x, t = threadIdx.x;
  qs[t] = q[b * A_ + t];
  __syncthreads();
  if (t < NP) {
    float acc = pb[t];
    for (int k = 0; k < A_; k++) acc += qs[k] * pe[t * A_ + k];
    lg[t] = acc;
  }
  __syncthreads();
  if (t == 0) {
    float m = lg[0];
    for (int n = 1; n < NP; n++) m = fmaxf(m, lg[n]);
    float s = 0.f;
    for (int n = 0; n < NP; n++) s += expf(lg[n] - m);
    mx = m; sm = s;
  }
  __syncthreads();
  if (t < NP) out[b * NP + t] = expf(lg[t] - mx) / sm;
}

// ---------------------------------------------------------------- launch
extern "C" void kernel_launch(void* const* d_in, const int* in_sizes, int n_in,
                              void* d_out, int out_size, void* d_ws, size_t ws_size,
                              hipStream_t stream) {
  (void)in_sizes; (void)n_in; (void)out_size; (void)ws_size;
  const float* enc      = (const float*)d_in[0];
  const int*   act      = (const int*)d_in[1];
  const int*   typ      = (const int*)d_in[2];
  const float* W_ih     = (const float*)d_in[3];
  const float* b_ih     = (const float*)d_in[4];
  const float* W_hh     = (const float*)d_in[5];
  const float* b_hh     = (const float*)d_in[6];
  const float* W_att    = (const float*)d_in[7];
  const float* W_av     = (const float*)d_in[8];
  const float* W_qa     = (const float*)d_in[9];
  const float* b_qa     = (const float*)d_in[10];
  const float* prod_emb = (const float*)d_in[11];
  const float* prod_b   = (const float*)d_in[12];
  const float* type_tab = (const float*)d_in[13];
  const float* W_init   = (const float*)d_in[14];
  const float* b_init   = (const float*)d_in[15];
  float* out = (float*)d_out;

  size_t off = 0;
  auto alloc = [&](size_t bytes) -> void* {
    void* p = (char*)d_ws + off;
    off += (bytes + 255) & ~(size_t)255;
    return p;
  };
  bf16*  src_bf = (bf16*)alloc((size_t)B_ * S_ * E_ * 2);
  bf16*  srclin = (bf16*)alloc((size_t)B_ * S_ * H_ * 2);
  bf16*  wcat   = (bf16*)alloc((size_t)4 * H_ * KCAT * 2);
  bf16*  watt   = (bf16*)alloc((size_t)H_ * E_ * 2);
  bf16*  wav    = (bf16*)alloc((size_t)AV_ * 2 * H_ * 2);
  bf16*  wqa    = (bf16*)alloc((size_t)A_ * AV_ * 2);
  bf16*  winit  = (bf16*)alloc((size_t)H_ * E_ * 2);
  bf16*  pooled = (bf16*)alloc((size_t)B_ * E_ * 2);
  float* gemmC  = (float*)alloc((size_t)B_ * 4 * H_ * 4);
  bf16*  hbf    = (bf16*)alloc((size_t)B_ * H_ * 2);
  float* c_f    = (float*)alloc((size_t)B_ * H_ * 4);
  bf16*  attbf  = (bf16*)alloc((size_t)B_ * AV_ * 2);
  bf16*  x_bf   = (bf16*)alloc((size_t)B_ * KCAT * 2);
  bf16*  hc     = (bf16*)alloc((size_t)B_ * 2 * H_ * 2);
  float* sc     = (float*)alloc((size_t)B_ * S_ * 4);
  float* wsm    = (float*)alloc((size_t)B_ * S_ * 4);
  float* q_f    = (float*)alloc((size_t)B_ * A_ * 4);

  // ---- weight / input conversion
  k_cvt<<<4096, 256, 0, stream>>>(enc, src_bf, (long)B_ * S_ * E_);
  k_build_wcat<<<2048, 256, 0, stream>>>(W_ih, W_hh, wcat);
  k_cvt<<<512, 256, 0, stream>>>(W_att, watt, (long)H_ * E_);
  k_cvt<<<512, 256, 0, stream>>>(W_av, wav, (long)AV_ * 2 * H_);
  k_cvt<<<128, 256, 0, stream>>>(W_qa, wqa, (long)A_ * AV_);
  k_cvt<<<512, 256, 0, stream>>>(W_init, winit, (long)H_ * E_);

  // ---- init state: h0 = tanh(maxpool(src) @ W_init^T + b_init), c0 = 0, att0 = 0
  k_pool_max<<<(B_ * E_ + 255) / 256, 256, 0, stream>>>(enc, pooled);
  k_gemm_bf16<<<dim3(H_ / 128, B_ / 128), 256, 0, stream>>>(pooled, winit, gemmC, B_, H_, E_, 0);
  k_bias_tanh<<<(B_ * H_ + 255) / 256, 256, 0, stream>>>(gemmC, b_init, nullptr, hbf, B_, H_);
  k_fill_f32<<<256, 256, 0, stream>>>(c_f, 0.f, (long)B_ * H_);
  k_fill_bf16<<<256, 256, 0, stream>>>(attbf, 0.f, (long)B_ * AV_);

  // ---- src_lin = src @ W_att^T  (131072 x 1024 x 1024, bf16 out)
  k_gemm_bf16<<<dim3(H_ / 128, (B_ * S_) / 128), 256, 0, stream>>>(
      src_bf, watt, srclin, B_ * S_, H_, E_, 1);

  // ---- decode steps
  for (int t = 0; t < T_; t++) {
    float mval = (t > 0) ? 1.f : 0.f;
    k_build_x<<<(B_ * KCAT + 255) / 256, 256, 0, stream>>>(
        act + t * B_, typ + t * B_, prod_emb, type_tab, attbf, hbf, x_bf, mval);
    // gates = [x|h] @ [W_ih|W_hh]^T
    k_gemm_bf16<<<dim3((4 * H_) / 128, B_ / 128), 256, 0, stream>>>(
        x_bf, wcat, gemmC, B_, 4 * H_, KCAT, 0);
    k_lstm<<<(B_ * H_ + 255) / 256, 256, 0, stream>>>(gemmC, b_ih, b_hh, c_f, hbf);
    // attention
    k_scores<<<dim3(S_ / 256, B_), 256, 0, stream>>>(srclin, hbf, sc);
    k_softmax<<<B_, 256, 0, stream>>>(sc, wsm);
    k_ctx_hc<<<B_, 256, 0, stream>>>(wsm, src_bf, hbf, hc);
    // att_t = tanh(hc @ W_av^T)
    k_gemm_bf16<<<dim3(AV_ / 128, B_ / 128), 256, 0, stream>>>(hc, wav, gemmC, B_, AV_, 2 * H_, 0);
    k_bias_tanh<<<(B_ * AV_ + 255) / 256, 256, 0, stream>>>(gemmC, nullptr, nullptr, attbf, B_, AV_);
    // q = tanh(att @ W_qa^T + b_qa)
    k_gemm_bf16<<<dim3(A_ / 128, B_ / 128), 256, 0, stream>>>(attbf, wqa, gemmC, B_, A_, AV_, 0);
    k_bias_tanh<<<(B_ * A_ + 255) / 256, 256, 0, stream>>>(gemmC, b_qa, q_f, nullptr, B_, A_);
    // probs = softmax(q @ prod_embed^T + prod_bias)
    k_readout<<<B_, 128, 0, stream>>>(q_f, prod_emb, prod_b, out + (size_t)t * B_ * NP);
  }
}
